// PINN_Raissi_70403103916493
// MI455X (gfx1250) — compile-verified, matrix operand order checked by reading, not run on hardware
//
#include <hip/hip_runtime.h>
#include <hip/hip_bf16.h>

// ---------------------------------------------------------------------------
// Types for CDNA5 WMMA (bf16 in, f32 accumulate)
// ---------------------------------------------------------------------------
typedef __bf16 bf16_t;
typedef __attribute__((ext_vector_type(16))) __bf16 v16bf;
typedef __attribute__((ext_vector_type(8)))  __bf16 v8bf;
typedef __attribute__((ext_vector_type(8)))  float  v8f;

union ABFrag { v16bf v; v8bf h[2]; };

#define HID 256
#define NMID 9
#define NPTS 8192

// ---------------------------------------------------------------------------
// Weight packing: f32 W[l][k][n] (row-major, k=in, n=out) -> bf16 fragments
// laid out so each lane of a wave reads its 16 halfs of the B operand of
// v_wmma_f32_16x16x32_bf16 as two contiguous 16-byte chunks.
// B (32x16, KxN) per-lane layout (mirror of documented 16-bit A layout):
//   lane = (n&15) | ((k>>3)&1)<<4 ;  half = (k&7) | ((k>>4)&1)<<3
// Fragment id: (l*8 + ks)*16 + nt   (ks = k/32, nt = n/16), 512 halfs each.
// ---------------------------------------------------------------------------
__global__ __launch_bounds__(256) void pack_w(const float* __restrict__ W,
                                              bf16_t* __restrict__ out) {
  int idx = blockIdx.x * 256 + threadIdx.x;           // l*65536 + k*256 + n
  if (idx >= NMID * HID * HID) return;
  int l = idx >> 16;
  int rem = idx & 65535;
  int k = rem >> 8;
  int n = rem & 255;
  int ks = k >> 5, nt = n >> 4;
  int lane = (n & 15) | (((k >> 3) & 1) << 4);
  int hi   = (k & 7)  | (((k >> 4) & 1) << 3);
  int dst = ((((l * 8 + ks) * 16 + nt) * 32 + lane) << 4) + hi;
  out[dst] = (bf16_t)W[idx];
}

// ---------------------------------------------------------------------------
// softplus100 Taylor-jet activation (Faa di Bruno to 3rd order).
// channels: 0=val 1=ax 2=ay 3=axx 4=axy 5=ayy 6=axxx 7=axxy 8=axyy 9=ayyy
// ---------------------------------------------------------------------------
template <int CH>
__device__ __forceinline__ void jet(const float* a, float* h) {
  float t   = 100.f * a[0];
  float sig = 1.f / (1.f + __expf(-t));
  float sp  = (t > 20.f) ? a[0] : __logf(1.f + __expf(t)) * 0.01f;
  float s1 = sig;
  h[0] = sp;
  h[1] = s1 * a[1];
  h[2] = s1 * a[2];
  if constexpr (CH == 10) {
    float s2 = 100.f * sig * (1.f - sig);
    float s3 = 100.f * s2 * (1.f - 2.f * sig);
    float ax = a[1], ay = a[2], axx = a[3], axy = a[4], ayy = a[5];
    h[3] = s2 * ax * ax + s1 * a[3];
    h[4] = s2 * ax * ay + s1 * a[4];
    h[5] = s2 * ay * ay + s1 * a[5];
    h[6] = s3 * ax * ax * ax + 3.f * s2 * axx * ax + s1 * a[6];
    h[7] = s3 * ax * ax * ay + s2 * (axx * ay + 2.f * axy * ax) + s1 * a[7];
    h[8] = s3 * ax * ay * ay + s2 * (ayy * ax + 2.f * axy * ay) + s1 * a[8];
    h[9] = s3 * ay * ay * ay + 3.f * s2 * ayy * ay + s1 * a[9];
  }
}

// ---------------------------------------------------------------------------
// Fused per-block MLP with derivative-jet propagation.
// CH channels per point, P points per block, ROWS = CH*P = 16*MT.
// Dynamic LDS: ROWS*256 floats; bf16 h-buffer aliases its low half
// (safe: every reuse is separated by read-all / barrier / write-all).
// 256 threads = 8 waves; wave w owns output columns [32w, 32w+32).
// ---------------------------------------------------------------------------
template <int CH, int P, int MT>
__global__ __launch_bounds__(256) void mlp_jet(
    const float* __restrict__ x, const float* __restrict__ y,
    const float* __restrict__ W0, const float* __restrict__ b0,
    const bf16_t* __restrict__ packedWm, const float* __restrict__ bm,
    const float* __restrict__ Wl, const float* __restrict__ bl,
    float* __restrict__ out) {
  constexpr int ROWS = CH * P;
  extern __shared__ char smem[];
  float*  zbuf = (float*)smem;        // [ROWS][256] f32 pre-activations
  bf16_t* hbuf = (bf16_t*)smem;       // [ROWS][256] bf16 activations (aliases)

  const int tid    = threadIdx.x;
  const int wave   = tid >> 5;
  const int lane   = tid & 31;
  const int lane15 = lane & 15;
  const int laneHi = lane >> 4;

  // ---- input layer: a = pt @ W0 + b0 ; jet through activation -------------
  float xv[P], yv[P];
#pragma unroll
  for (int pp = 0; pp < P; ++pp) {
    xv[pp] = x[blockIdx.x * P + pp];
    yv[pp] = y[blockIdx.x * P + pp];
  }
  {
    const int n = tid;                // neuron = thread
    const float w0n = W0[n], w1n = W0[HID + n], b0n = b0[n];
#pragma unroll
    for (int pp = 0; pp < P; ++pp) {
      float a[10] = {xv[pp] * w0n + yv[pp] * w1n + b0n,
                     w0n, w1n, 0.f, 0.f, 0.f, 0.f, 0.f, 0.f, 0.f};
      float h[CH];
      jet<CH>(a, h);
#pragma unroll
      for (int c = 0; c < CH; ++c)
        hbuf[(pp * CH + c) * 256 + n] = (bf16_t)h[c];
    }
  }
  __syncthreads();

  // ---- 9 hidden layers: WMMA GEMM [ROWS,256]x[256,256] + jet --------------
  for (int l = 0; l < NMID; ++l) {
    v8f acc[MT][2];
#pragma unroll
    for (int m = 0; m < MT; ++m) {
      acc[m][0] = (v8f)(0.0f);
      acc[m][1] = (v8f)(0.0f);
    }
#pragma unroll
    for (int ks = 0; ks < 8; ++ks) {
      // B fragments for this wave's two 16-column tiles (L2-resident).
      ABFrag bf[2];
#pragma unroll
      for (int ns = 0; ns < 2; ++ns) {
        const bf16_t* bp = packedWm +
            (((((l * 8 + ks) * 16) + (wave * 2 + ns)) * 32 + lane) << 4);
        bf[ns].h[0] = *(const v8bf*)(bp);
        bf[ns].h[1] = *(const v8bf*)(bp + 8);
      }
#pragma unroll
      for (int m = 0; m < MT; ++m) {
        // A fragment from LDS: row = m*16+lane15, K-chunks per 16-bit layout.
        const int base = (m * 16 + lane15) * 256 + ks * 32 + 8 * laneHi;
        ABFrag af;
        af.h[0] = *(const v8bf*)(hbuf + base);
        af.h[1] = *(const v8bf*)(hbuf + base + 16);
        acc[m][0] = __builtin_amdgcn_wmma_f32_16x16x32_bf16(
            false, af.v, false, bf[0].v, (short)0, acc[m][0], false, false);
        acc[m][1] = __builtin_amdgcn_wmma_f32_16x16x32_bf16(
            false, af.v, false, bf[1].v, (short)0, acc[m][1], false, false);
      }
    }
    __syncthreads();  // all waves done reading hbuf

    // store z (bias only on the value channel, row % CH == 0)
    const float* bmL = bm + l * HID;
#pragma unroll
    for (int m = 0; m < MT; ++m)
#pragma unroll
      for (int ns = 0; ns < 2; ++ns) {
        const int col = wave * 32 + ns * 16 + lane15;
#pragma unroll
        for (int r = 0; r < 8; ++r) {
          const int row = m * 16 + r + 8 * laneHi;
          float zv = acc[m][ns][r];
          if (row % CH == 0) zv += bmL[col];
          zbuf[row * 256 + col] = zv;
        }
      }
    __syncthreads();  // z complete

    // activation: read-all -> barrier -> write-all (zbuf/hbuf alias-safe)
    float zz[P][CH];
    {
      const int n = tid;
#pragma unroll
      for (int pp = 0; pp < P; ++pp)
#pragma unroll
        for (int c = 0; c < CH; ++c)
          zz[pp][c] = zbuf[(pp * CH + c) * 256 + n];
    }
    __syncthreads();
    {
      const int n = tid;
#pragma unroll
      for (int pp = 0; pp < P; ++pp) {
        float h[CH];
        jet<CH>(zz[pp], h);
#pragma unroll
        for (int c = 0; c < CH; ++c)
          hbuf[(pp * CH + c) * 256 + n] = (bf16_t)h[c];
      }
    }
    __syncthreads();
  }

  // ---- output layer: per-row dot with Wl (256 -> 1) -----------------------
  for (int row = tid; row < ROWS; row += 256) {
    float s = 0.f;
    for (int nn = 0; nn < 256; ++nn) {
      int n = (nn + row * 8) & 255;  // skew to dodge LDS bank conflicts
      s += (float)hbuf[row * 256 + n] * Wl[n];
    }
    int pt = blockIdx.x * P + row / CH;
    int c = row % CH;
    out[pt * CH + c] = s + (c == 0 ? bl[0] : 0.f);
  }
}

// ---------------------------------------------------------------------------
// Physics residual + 9-way masked/unmasked sum reduction (wave shuffle +
// f32 global atomics).
// ---------------------------------------------------------------------------
__global__ __launch_bounds__(256) void physics_reduce(
    const float* __restrict__ psiO, const float* __restrict__ pO,
    const float* __restrict__ u, const float* __restrict__ v,
    const float* __restrict__ pr, const unsigned char* __restrict__ mask,
    float* __restrict__ acc) {
  const int i = blockIdx.x * 256 + threadIdx.x;
  const float* po = psiO + i * 10;
  float gx = po[1], gy = po[2];
  float hxx = po[3], hxy = po[4], hyy = po[5];
  float txxx = po[6], txxy = po[7], txyy = po[8], tyyy = po[9];
  float pv = pO[i * 3 + 0], px = pO[i * 3 + 1], py = pO[i * 3 + 2];

  float up = gy, vp = -gx;
  float ux = hxy, uy = hyy, vx = -hxx, vy = -hxy;
  float uxx = txxy, uyy = tyyy, vxx = -txxx, vyy = -txyy;
  float fu = 2.f * up * ux + up * vy + vp * uy + px - 0.01f * (uxx + uyy);
  float fv = 2.f * vp * vy + up * vx + vp * ux + py - 0.01f * (vxx + vyy);

  float ut = u[i], vt = v[i], pt = pr[i];
  float m = mask[i] ? 1.f : 0.f;
  float du = ut - up, dv = vt - vp, dp = pt - pv;

  float vals[9] = {m, m * du * du, m * dv * dv, m * dp * dp,
                   du * du, dv * dv, dp * dp, fu * fu, fv * fv};
#pragma unroll
  for (int q = 0; q < 9; ++q) {
    float s = vals[q];
    for (int off = 16; off > 0; off >>= 1) s += __shfl_xor(s, off, 32);
    if ((threadIdx.x & 31) == 0) atomicAdd(&acc[q], s);
  }
}

__global__ void zero_accum(float* acc) {
  if (threadIdx.x < 16) acc[threadIdx.x] = 0.f;
}

__global__ void finalize(const float* __restrict__ a, float* __restrict__ out) {
  float ms = fmaxf(a[0], 1.f);
  float invN = 1.f / (float)NPTS;
  float rans = (a[7] + a[8]) * invN;
  out[0] = a[1] / ms + a[2] / ms + rans;  // psi_loss
  out[1] = a[3] / ms + rans;              // p_loss
  out[2] = a[4] * invN;                   // u_tr
  out[3] = a[5] * invN;                   // v_tr
  out[4] = a[6] * invN;                   // p_tr
  out[5] = rans;                          // rans
}

// ---------------------------------------------------------------------------
// Launch
// ---------------------------------------------------------------------------
extern "C" void kernel_launch(void* const* d_in, const int* in_sizes, int n_in,
                              void* d_out, int out_size, void* d_ws,
                              size_t ws_size, hipStream_t stream) {
  const float* x  = (const float*)d_in[0];
  const float* y  = (const float*)d_in[1];
  const float* u  = (const float*)d_in[2];
  const float* v  = (const float*)d_in[3];
  const float* p  = (const float*)d_in[4];
  const unsigned char* mask = (const unsigned char*)d_in[5];
  const float* psi_W0 = (const float*)d_in[6];
  const float* psi_b0 = (const float*)d_in[7];
  const float* psi_Wm = (const float*)d_in[8];
  const float* psi_bm = (const float*)d_in[9];
  const float* psi_Wl = (const float*)d_in[10];
  const float* psi_bl = (const float*)d_in[11];
  const float* p_W0 = (const float*)d_in[12];
  const float* p_b0 = (const float*)d_in[13];
  const float* p_Wm = (const float*)d_in[14];
  const float* p_bm = (const float*)d_in[15];
  const float* p_Wl = (const float*)d_in[16];
  const float* p_bl = (const float*)d_in[17];

  char* ws = (char*)d_ws;
  const size_t packBytes = (size_t)NMID * HID * HID * sizeof(bf16_t);  // 1.125 MB
  bf16_t* psiPack = (bf16_t*)(ws);
  bf16_t* pPack   = (bf16_t*)(ws + packBytes);
  float*  psiOut  = (float*)(ws + 2 * packBytes);                      // 8192*10
  float*  pOut    = (float*)(ws + 2 * packBytes + NPTS * 10 * sizeof(float));
  float*  accum   = (float*)(ws + 2 * packBytes + NPTS * 13 * sizeof(float));

  const int packBlocks = (NMID * HID * HID + 255) / 256;
  pack_w<<<packBlocks, 256, 0, stream>>>(psi_Wm, psiPack);
  pack_w<<<packBlocks, 256, 0, stream>>>(p_Wm, pPack);
  zero_accum<<<1, 32, 0, stream>>>(accum);

  // psi net: 10 jet channels, 8 points/block -> 80 rows (5 M-tiles), 80KB LDS
  mlp_jet<10, 8, 5><<<NPTS / 8, 256, 80 * 256 * sizeof(float), stream>>>(
      x, y, psi_W0, psi_b0, psiPack, psi_bm, psi_Wl, psi_bl, psiOut);
  // p net: 3 jet channels, 16 points/block -> 48 rows (3 M-tiles), 48KB LDS
  mlp_jet<3, 16, 3><<<NPTS / 16, 256, 48 * 256 * sizeof(float), stream>>>(
      x, y, p_W0, p_b0, pPack, p_bm, p_Wl, p_bl, pOut);

  physics_reduce<<<NPTS / 256, 256, 0, stream>>>(psiOut, pOut, u, v, p, mask,
                                                 accum);
  finalize<<<1, 1, 0, stream>>>(accum, (float*)d_out);
}
